// EncoderPredictor_1864015806866
// MI455X (gfx1250) — matrix operand order repaired and checked by reading, not compiled, not run on hardware
//
#include <hip/hip_runtime.h>
#include <hip/hip_bf16.h>

// Problem dims (compile-time constants from the reference)
#define BQ    64      // batch (queries)
#define KK    4096    // keys
#define DD    512     // embed dim
#define HH    512     // hidden dim

typedef __attribute__((ext_vector_type(2)))  float v2f;
typedef __attribute__((ext_vector_type(8)))  float v8f;

#if defined(__has_builtin)
#if __has_builtin(__builtin_amdgcn_wmma_f32_16x16x4_f32)
#define USE_WMMA_F32 1
#endif
#endif

#ifndef USE_WMMA_F32
typedef __attribute__((ext_vector_type(16))) __bf16 v16bf;
#endif

// ---------------------------------------------------------------------------
// Stage 1/2: C[M x 512] = A[M x 512] @ W[512 x 512] (+ bias), fp32 WMMA.
// One wave32 computes one 16x16 C tile; K loop = 128 x v_wmma_f32_16x16x4_f32.
// If idx != nullptr, A rows are gathered through idx (embedding lookup).
// ---------------------------------------------------------------------------
__global__ void gemm_wmma(const float* __restrict__ A,
                          const int*   __restrict__ idx,
                          const float* __restrict__ W,
                          const float* __restrict__ bias,
                          float*       __restrict__ C,
                          int M)
{
    const int lane = threadIdx.x & 31;
    const int wave = threadIdx.x >> 5;
    const int waves_per_block = blockDim.x >> 5;

    const int tiles_n = HH / 16;                 // 32
    const int tile    = blockIdx.x * waves_per_block + wave;
    const int total   = (M / 16) * tiles_n;
    if (tile >= total) return;

    const int mt = tile / tiles_n;
    const int nt = tile % tiles_n;

    const int r    = lane & 15;                  // row within tile (A), col (B/C/D)
    const int half = lane >> 4;                  // lane half selects K pair / M half

    const int arow_logical = mt * 16 + r;
    const int arow = idx ? idx[arow_logical] : arow_logical;
    const float* __restrict__ arow_ptr = A + (size_t)arow * DD;
    const int n0 = nt * 16;

    v8f acc = {};

#ifdef USE_WMMA_F32
    // A 16x4 fp32: v0 = A[r, 2*half], v1 = A[r, 2*half+1]  (ISA 7.12.2)
    // B 4x16 fp32: v0 = W[2*half,    n0+r], v1 = W[2*half+1, n0+r]
    for (int kk = 0; kk < DD; kk += 4) {
        v2f a = *(const v2f*)(arow_ptr + kk + 2 * half);
        v2f b;
        b.x = W[(kk + 2 * half    ) * HH + n0 + r];
        b.y = W[(kk + 2 * half + 1) * HH + n0 + r];
        acc = __builtin_amdgcn_wmma_f32_16x16x4_f32(
            /*neg_a=*/false, a, /*neg_b=*/false, b,
            /*c_mod=*/(short)0, acc, /*reuse_a=*/false, /*reuse_b=*/false);
    }
#else
    // Fallback: probe-confirmed bf16 WMMA (16x16x32), fp32 accumulate.
    for (int kk = 0; kk < DD; kk += 32) {
        v16bf a, b;
#pragma unroll
        for (int e = 0; e < 16; ++e) {
            const int j  = e >> 1;
            const int ka = ((j & 3) * 2 + (e & 1)) + ((j >> 2) * 16) + half * 8;
            a[e] = (__bf16)arow_ptr[kk + ka];
            const int kb = (e & 7) + ((e >> 3) * 16) + half * 8;
            b[e] = (__bf16)W[(kk + kb) * HH + n0 + r];
        }
        acc = __builtin_amdgcn_wmma_f32_16x16x32_bf16(
            false, a, false, b, (short)0, acc, false, false);
    }
#endif

    // D 16x16 fp32: VGPR i -> row half*8+i, col r   (ISA 7.12.2)
    const float bv = bias ? bias[n0 + r] : 0.0f;
#pragma unroll
    for (int i = 0; i < 8; ++i) {
        const int m = mt * 16 + half * 8 + i;
        C[(size_t)m * HH + n0 + r] = acc[i] + bv;
    }
}

// ---------------------------------------------------------------------------
// Stage 3: logits[b,k] = sum_h w2[h] * relu(qh'[b,h] + kh[k,h]) + b2
// (b1 already folded into qh'). Block = 128 threads = 16 b x 8 k outputs.
//
// LDS staging uses GLOBAL_LOAD_ASYNC_TO_LDS_B128 (ASYNCcnt path, CDNA5):
// data goes global->LDS without touching VGPRs; fenced with s_wait_asynccnt.
// sq rows padded to 516 floats: 16B-aligned rows, and for the float4 compute
// reads word-bank = (4*b + h) % 64 -> each half-wave covers all 64 banks once.
// kh/w2 reads are half-wave/fully uniform (LDS broadcast).
// ---------------------------------------------------------------------------
#define BT 16
#define KT 8
#define SQS (DD + 4)          // 516-float padded row stride

__global__ void score_kernel(const float* __restrict__ qh,
                             const float* __restrict__ kh,
                             const float* __restrict__ w2,
                             const float* __restrict__ b2,
                             float*       __restrict__ out)
{
    __shared__ __align__(16) float sq[BT][SQS];
    __shared__ __align__(16) float sk[KT][HH];
    __shared__ __align__(16) float sw[HH];

    const int tid = threadIdx.x;                 // 128 threads
    const int b0  = (blockIdx.x & 3) * BT;       // 4 b-tiles
    const int k0  = (blockIdx.x >> 2) * KT;      // 512 k-tiles

    // Generic LDS pointers -> 32-bit LDS byte offsets (addr[31:0] per ISA 10.2)
    const unsigned sq_base = (unsigned)(uintptr_t)&sq[0][0];
    const unsigned sk_base = (unsigned)(uintptr_t)&sk[0][0];
    const unsigned sw_base = (unsigned)(uintptr_t)&sw[0];

    // qh' tile: 16 rows x 512 floats -> padded LDS rows (128 x b128 / thread set)
    for (int i = tid; i < BT * (DD / 4); i += 128) {
        const int bb = i >> 7;                   // / (DD/4)
        const int h4 = i & (DD / 4 - 1);
        const unsigned lds = sq_base + (unsigned)(bb * SQS + h4 * 4) * 4u;
        const unsigned off = (unsigned)(((b0 + bb) * DD + h4 * 4) * 4);
        asm volatile("global_load_async_to_lds_b128 %0, %1, %2"
                     :: "v"(lds), "v"(off), "s"(qh) : "memory");
    }
    // kh tile: contiguous 8 x 512 floats
    for (int i = tid; i < KT * (HH / 4); i += 128) {
        const unsigned lds = sk_base + (unsigned)i * 16u;
        const unsigned off = (unsigned)((k0 * HH + i * 4) * 4);
        asm volatile("global_load_async_to_lds_b128 %0, %1, %2"
                     :: "v"(lds), "v"(off), "s"(kh) : "memory");
    }
    // w2: 512 floats = 128 x b128
    {
        const unsigned lds = sw_base + (unsigned)tid * 16u;
        const unsigned off = (unsigned)tid * 16u;
        asm volatile("global_load_async_to_lds_b128 %0, %1, %2"
                     :: "v"(lds), "v"(off), "s"(w2) : "memory");
    }
    asm volatile("s_wait_asynccnt 0" ::: "memory");
    __syncthreads();

    const int bb = tid & 15;
    const int kq = tid >> 4;

    const float4* __restrict__ q4 = (const float4*)&sq[bb][0];
    const float4* __restrict__ k4 = (const float4*)&sk[kq][0];
    const float4* __restrict__ w4 = (const float4*)&sw[0];

    float a0 = 0.f, a1 = 0.f, a2 = 0.f, a3 = 0.f;
#pragma unroll 4
    for (int i = 0; i < HH / 4; ++i) {
        const float4 q = q4[i];
        const float4 k = k4[i];
        const float4 w = w4[i];
        a0 = __builtin_fmaf(w.x, fmaxf(q.x + k.x, 0.f), a0);
        a1 = __builtin_fmaf(w.y, fmaxf(q.y + k.y, 0.f), a1);
        a2 = __builtin_fmaf(w.z, fmaxf(q.z + k.z, 0.f), a2);
        a3 = __builtin_fmaf(w.w, fmaxf(q.w + k.w, 0.f), a3);
    }
    out[(size_t)(b0 + bb) * KK + (k0 + kq)] = (a0 + a1) + (a2 + a3) + b2[0];
}

// ---------------------------------------------------------------------------
// Launch: 3 kernels, all on `stream`. Workspace: qh' (64x512 f32) then
// kh (4096x512 f32) — everything stays L2-resident (192 MB L2).
// ---------------------------------------------------------------------------
extern "C" void kernel_launch(void* const* d_in, const int* in_sizes, int n_in,
                              void* d_out, int out_size, void* d_ws, size_t ws_size,
                              hipStream_t stream)
{
    const int*   Xq    = (const int*)  d_in[0];               // (64,1) int32
    const float* qemb  = (const float*)d_in[1];               // (8192,512)
    const float* kemb  = (const float*)d_in[2];               // (4096,512)
    const float* W1    = (const float*)d_in[3];               // (1024,512)
    const float* b1    = (const float*)d_in[4];               // (512,)
    const float* W2    = (const float*)d_in[5];               // (512,1)
    const float* b2    = (const float*)d_in[6];               // (1,)
    float*       out   = (float*)d_out;                       // (64,4096)

    float* qh_ws = (float*)d_ws;                              // 64*512 f32
    float* kh_ws = qh_ws + (size_t)BQ * HH;                   // 4096*512 f32

    // Stage 1: qh' = gather(qemb, Xq) @ W1[:512] + b1  -> 128 tiles, 8 waves/block
    gemm_wmma<<<16, 256, 0, stream>>>(qemb, Xq, W1, b1, qh_ws, BQ);

    // Stage 2: kh = kemb @ W1[512:]                    -> 8192 tiles
    gemm_wmma<<<1024, 256, 0, stream>>>(kemb, nullptr, W1 + (size_t)DD * HH,
                                        nullptr, kh_ws, KK);

    // Stage 3: fused relu-dot score                    -> 4 x 512 tiles
    score_kernel<<<2048, 128, 0, stream>>>(qh_ws, kh_ws, W2, b2, out);
}
